// ShiftedWindowAttention_16045997818172
// MI455X (gfx1250) — matrix operand order repaired and checked by reading, not compiled, not run on hardware
//
#include <hip/hip_runtime.h>

#define DIMX 192
#define HEADS 6
#define HD 32
#define WSZ 7
#define LTOK 49
#define LP 64
#define HW 56
#define SHFT 3
#define QKSCALE 0.17677669529663687f

typedef __attribute__((ext_vector_type(16))) __bf16 v16bf;
typedef __attribute__((ext_vector_type(2)))  __bf16 v2bf;
typedef __attribute__((ext_vector_type(8)))  float  v8f;
typedef __attribute__((ext_vector_type(8)))  unsigned short u16x8;
typedef __attribute__((ext_vector_type(4)))  unsigned short u16x4;

union FragAB { v16bf v; u16x8 h[2]; };
union FragC  { v8f  v; float f[8]; };

__device__ inline unsigned short f2bf(float f) {
  unsigned u = __builtin_bit_cast(unsigned, f);
  unsigned r = u + 0x7FFFu + ((u >> 16) & 1u);
  return (unsigned short)(r >> 16);
}

#if defined(__has_builtin)
#if __has_builtin(__builtin_amdgcn_cvt_pk_bf16_f32)
#define HAVE_PK_BF16 1
#endif
#endif

// pack two f32 -> two bf16 in one 32-bit value (lo = a, hi = b)
// gfx1250 has V_CVT_PK_BF16_F32 but clang doesn't declare the builtin -> inline asm
__device__ inline unsigned pk2bf(float a, float b) {
#ifdef HAVE_PK_BF16
  v2bf r = __builtin_amdgcn_cvt_pk_bf16_f32(a, b);
  return __builtin_bit_cast(unsigned, r);
#else
  unsigned r;
  asm("v_cvt_pk_bf16_f32 %0, %1, %2" : "=v"(r) : "v"(a), "v"(b));
  return r;
#endif
}

__device__ inline int rel_index(int i, int j) {
  int y1 = i / WSZ, x1 = i % WSZ, y2 = j / WSZ, x2 = j % WSZ;
  return (y1 - y2 + WSZ - 1) * (2 * WSZ - 1) + (x1 - x2 + WSZ - 1);
}

// A-fragment (16-bit A 16x32 ISA table): lane(ln,grp), elem h -> K = h + (h&8) + 8*grp
__device__ inline FragAB loadFragA(const unsigned short* rowBase, int grp) {
  FragAB f;
  const u16x8* p = (const u16x8*)rowBase;
  f.h[0] = p[grp];
  f.h[1] = p[2 + grp];
  return f;
}
// B-fragment (K x 16 striping): lane(ln,grp), elem h -> K = h + (h&8) + 8*grp (matches A)
__device__ inline FragAB loadFragB(const unsigned short* rowBase, int grp) {
  FragAB f;
  const u16x8* p = (const u16x8*)rowBase;
  f.h[0] = p[grp];
  f.h[1] = p[2 + grp];
  return f;
}

// ---------------- pre-pass: fp32 weights -> bf16 in workspace ----------------
__global__ __launch_bounds__(256)
void convert_weights(const float* __restrict__ w_qkv, const float* __restrict__ w_out,
                     unsigned short* __restrict__ wq_bf, unsigned short* __restrict__ wo_bf) {
  const int NQ = 3 * DIMX * DIMX;   // 110592
  const int NO = DIMX * DIMX;       // 36864
  int idx = (blockIdx.x * 256 + threadIdx.x) * 4;
  if (idx < NQ) {
    float4 v = *(const float4*)&w_qkv[idx];
    unsigned lo = pk2bf(v.x, v.y), hi = pk2bf(v.z, v.w);
    u16x4 o = { (unsigned short)lo, (unsigned short)(lo >> 16),
                (unsigned short)hi, (unsigned short)(hi >> 16) };
    *(u16x4*)&wq_bf[idx] = o;
  } else if (idx < NQ + NO) {
    int k = idx - NQ;
    float4 v = *(const float4*)&w_out[k];
    unsigned lo = pk2bf(v.x, v.y), hi = pk2bf(v.z, v.w);
    u16x4 o = { (unsigned short)lo, (unsigned short)(lo >> 16),
                (unsigned short)hi, (unsigned short)(hi >> 16) };
    *(u16x4*)&wo_bf[k] = o;
  }
}

// ---------------- fused shifted-window attention, one block per window ----------------
__global__ __launch_bounds__(256)
void swin_attn_fused(const float* __restrict__ x,
                     const unsigned short* __restrict__ wq_bf,
                     const float* __restrict__ b_qkv,
                     const unsigned short* __restrict__ wo_bf,
                     const float* __restrict__ b_out,
                     const float* __restrict__ pos_enc,
                     float* __restrict__ out)
{
  constexpr int QK_OFF  = 0;                           // ushort qk[64][384]   (q | k)
  constexpr int VT_OFF  = LP * 2 * DIMX * 2;           // ushort vT[192][64]   (v transposed)
  constexpr int XS_OFF  = VT_OFF + DIMX * LP * 2;      // ushort xA[64][192]  OR  scores+probs
  constexpr int AO_OFF  = XS_OFF + LP * DIMX * 2;      // ushort attnO[64][192]
  constexpr int TOK_OFF = AO_OFF + LP * DIMX * 2;      // int[64]
  constexpr int REG_OFF = TOK_OFF + LP * 4;            // int[64]
  constexpr int REL_OFF = REG_OFF + LP * 4;            // u8[49*49], pad
  __shared__ __align__(16) unsigned char smem[REL_OFF + 2432];

  auto qk     = (unsigned short (*)[2 * DIMX])(smem + QK_OFF);
  auto vT     = (unsigned short (*)[LP])(smem + VT_OFF);
  auto xA     = (unsigned short (*)[DIMX])(smem + XS_OFF);
  auto scores = (float (*)[LP])(smem + XS_OFF);
  auto probs  = (unsigned short (*)[LP])(smem + XS_OFF + LP * LP * 4);
  auto attnO  = (unsigned short (*)[DIMX])(smem + AO_OFF);
  int* tokOff = (int*)(smem + TOK_OFF);
  int* regin  = (int*)(smem + REG_OFF);
  unsigned char* relIdxS = (unsigned char*)(smem + REL_OFF);

  const int tid  = threadIdx.x;
  const int wave = tid >> 5;
  const int lane = tid & 31;
  const int ln   = lane & 15;
  const int grp  = lane >> 4;

  const int blk  = blockIdx.x;
  const int bimg = blk >> 6;
  const int win  = blk & 63;
  const int wy = win >> 3, wx = win & 7;
  const long xbase = (long)bimg * (HW * HW) * DIMX;

  if (tid < LP) {
    int t = tid;
    if (t < LTOK) {
      int ly = t / WSZ, lx = t % WSZ;
      int gy = wy * WSZ + ly, gx = wx * WSZ + lx;
      int y  = (gy + SHFT) % HW, xx = (gx + SHFT) % HW;
      tokOff[t] = (y * HW + xx) * DIMX;
      int ry = gy < (HW - WSZ) ? 0 : (gy < (HW - SHFT) ? 1 : 2);
      int rx = gx < (HW - WSZ) ? 0 : (gx < (HW - SHFT) ? 1 : 2);
      regin[t] = ry * 3 + rx;
    } else { tokOff[t] = -1; regin[t] = -9; }
  }
  for (int idx = tid; idx < LTOK * LTOK; idx += 256)
    relIdxS[idx] = (unsigned char)rel_index(idx / LTOK, idx % LTOK);
  __syncthreads();

  // gather window tokens -> LDS bf16, zero-pad rows 49..63
  for (int idx = tid; idx < LP * (DIMX / 4); idx += 256) {
    int row = idx / (DIMX / 4);
    int c4  = (idx % (DIMX / 4)) * 4;
    float4 v = make_float4(0.f, 0.f, 0.f, 0.f);
    if (row < LTOK) v = *(const float4*)&x[xbase + tokOff[row] + c4];
    unsigned lo = pk2bf(v.x, v.y), hi = pk2bf(v.z, v.w);
    u16x4 o = { (unsigned short)lo, (unsigned short)(lo >> 16),
                (unsigned short)hi, (unsigned short)(hi >> 16) };
    *(u16x4*)&xA[row][c4] = o;
  }
  __syncthreads();

  // ---------------- QKV projection ----------------
  // q gets the attention scale folded in here (removes a mul from softmax inner loop)
  for (int t = wave; t < 4 * 36; t += 8) {
    int mT = t / 36, nT = t % 36;
    const int ocol = nT * 16 + ln;
    FragC c;
    #pragma unroll
    for (int r = 0; r < 8; ++r) c.f[r] = 0.f;
    #pragma unroll
    for (int kc = 0; kc < 6; ++kc) {
      FragAB a = loadFragA(&xA[mT * 16 + ln][kc * 32], grp);
      FragAB b = loadFragB(&wq_bf[ocol * DIMX + kc * 32], grp);
      c.v = __builtin_amdgcn_wmma_f32_16x16x32_bf16(false, a.v, false, b.v,
                                                    (short)0, c.v, false, false);
    }
    float bias = b_qkv[ocol];
    float scl  = (nT < 12) ? QKSCALE : 1.0f;   // scale only q columns
    if (nT < 24) {                             // q or k -> row-major [tok][col]
      const int mrow = mT * 16 + (grp << 3);
      #pragma unroll
      for (int r = 0; r < 8; r += 2) {
        unsigned pk = pk2bf((c.f[r] + bias) * scl, (c.f[r + 1] + bias) * scl);
        qk[mrow + r][ocol]     = (unsigned short)pk;
        qk[mrow + r + 1][ocol] = (unsigned short)(pk >> 16);
      }
    } else {                                   // v -> transposed [d][tok], one 16B store
      u16x8 o;
      #pragma unroll
      for (int r = 0; r < 8; r += 2) {
        unsigned pk = pk2bf(c.f[r] + bias, c.f[r + 1] + bias);
        o[r]     = (unsigned short)pk;
        o[r + 1] = (unsigned short)(pk >> 16);
      }
      *(u16x8*)&vT[ocol - 2 * DIMX][mT * 16 + (grp << 3)] = o;
    }
  }
  __syncthreads();

  // ---------------- per-head attention ----------------
  for (int hh = 0; hh < HEADS; ++hh) {
    // scores = (q*scale) @ k^T : 4x4 tiles, K=32 -> one WMMA each
    for (int t = wave; t < 16; t += 8) {
      int mT = t >> 2, nT = t & 3;
      FragAB a = loadFragA(&qk[mT * 16 + ln][hh * HD], grp);
      FragAB b = loadFragB(&qk[nT * 16 + ln][DIMX + hh * HD], grp);
      FragC c;
      #pragma unroll
      for (int r = 0; r < 8; ++r) c.f[r] = 0.f;
      c.v = __builtin_amdgcn_wmma_f32_16x16x32_bf16(false, a.v, false, b.v,
                                                    (short)0, c.v, false, false);
      #pragma unroll
      for (int r = 0; r < 8; ++r)
        scores[mT * 16 + (grp << 3) + r][nT * 16 + ln] = c.f[r];
    }
    __syncthreads();

    // softmax (+rel bias, +shift mask), one row per thread
    if (tid < LP) {
      int i = tid;
      if (i < LTOK) {
        const float* pe = pos_enc + hh * 169;
        const unsigned char* rrow = relIdxS + i * LTOK;
        int ri = regin[i];
        float mx = -1e30f;
        for (int j = 0; j < LTOK; ++j) {
          float s = scores[i][j] + pe[rrow[j]];
          if (regin[j] != ri) s = -1e30f;
          scores[i][j] = s;
          mx = fmaxf(mx, s);
        }
        float sum = 0.f;
        for (int j = 0; j < LTOK; ++j) {
          float e = __expf(scores[i][j] - mx);
          scores[i][j] = e;
          sum += e;
        }
        float inv = 1.f / sum;
        for (int j = 0; j < LTOK - 1; j += 2) {
          unsigned pk = pk2bf(scores[i][j] * inv, scores[i][j + 1] * inv);
          probs[i][j]     = (unsigned short)pk;
          probs[i][j + 1] = (unsigned short)(pk >> 16);
        }
        probs[i][LTOK - 1] = f2bf(scores[i][LTOK - 1] * inv);
        for (int j = LTOK; j < LP; ++j) probs[i][j] = 0;
      } else {
        for (int j = 0; j < LP; ++j) probs[tid][j] = 0;
      }
    }
    __syncthreads();

    // out_h = probs @ v : 4x2 tiles (one per wave), K=64 -> 2 chained WMMAs
    {
      int mT = wave >> 1, nT = wave & 1;
      FragC c;
      #pragma unroll
      for (int r = 0; r < 8; ++r) c.f[r] = 0.f;
      #pragma unroll
      for (int kc = 0; kc < 2; ++kc) {
        FragAB a = loadFragA(&probs[mT * 16 + ln][kc * 32], grp);
        FragAB b = loadFragB(&vT[hh * HD + nT * 16 + ln][kc * 32], grp);
        c.v = __builtin_amdgcn_wmma_f32_16x16x32_bf16(false, a.v, false, b.v,
                                                      (short)0, c.v, false, false);
      }
      const int mrow = mT * 16 + (grp << 3);
      const int acol = hh * HD + nT * 16 + ln;
      #pragma unroll
      for (int r = 0; r < 8; r += 2) {
        unsigned pk = pk2bf(c.f[r], c.f[r + 1]);
        attnO[mrow + r][acol]     = (unsigned short)pk;
        attnO[mrow + r + 1][acol] = (unsigned short)(pk >> 16);
      }
    }
    __syncthreads();
  }

  // ---------------- output projection + un-shift scatter ----------------
  const long obase = (long)bimg * (HW * HW) * DIMX;
  for (int t = wave; t < 4 * 12; t += 8) {
    int mT = t / 12, nT = t % 12;
    const int ocol = nT * 16 + ln;
    FragC c;
    #pragma unroll
    for (int r = 0; r < 8; ++r) c.f[r] = 0.f;
    #pragma unroll
    for (int kc = 0; kc < 6; ++kc) {
      FragAB a = loadFragA(&attnO[mT * 16 + ln][kc * 32], grp);
      FragAB b = loadFragB(&wo_bf[ocol * DIMX + kc * 32], grp);
      c.v = __builtin_amdgcn_wmma_f32_16x16x32_bf16(false, a.v, false, b.v,
                                                    (short)0, c.v, false, false);
    }
    float bias = b_out[ocol];
    #pragma unroll
    for (int r = 0; r < 8; ++r) {
      int m = mT * 16 + (grp << 3) + r;
      if (m < LTOK) out[obase + tokOff[m] + ocol] = c.f[r] + bias;
    }
  }
}

extern "C" void kernel_launch(void* const* d_in, const int* in_sizes, int n_in,
                              void* d_out, int out_size, void* d_ws, size_t ws_size,
                              hipStream_t stream) {
  const float* x       = (const float*)d_in[0];
  const float* w_qkv   = (const float*)d_in[1];
  const float* b_qkv   = (const float*)d_in[2];
  const float* w_out   = (const float*)d_in[3];
  const float* b_out   = (const float*)d_in[4];
  const float* pos_enc = (const float*)d_in[5];
  float* out = (float*)d_out;
  (void)in_sizes; (void)n_in; (void)out_size; (void)ws_size;

  unsigned short* wq_bf = (unsigned short*)d_ws;                    // 3*192*192 bf16
  unsigned short* wo_bf = wq_bf + 3 * DIMX * DIMX;                  // 192*192 bf16

  const int NCVT = (3 * DIMX * DIMX + DIMX * DIMX) / 4;             // 36864 float4s
  convert_weights<<<(NCVT + 255) / 256, 256, 0, stream>>>(w_qkv, w_out, wq_bf, wo_bf);

  swin_attn_fused<<<32 * 64, 256, 0, stream>>>(x, wq_bf, b_qkv, wo_bf, b_out, pos_enc, out);
}
